// Decoder_80109730005608
// MI455X (gfx1250) — compile-verified, matrix operand order checked by reading, not run on hardware
//
#include <hip/hip_runtime.h>

// ---------------------------------------------------------------------------
// Edge-MLP decoder for MI455X (gfx1250, wave32, WMMA, async-to-LDS).
//
// 41 GFLOP total; x fits in the 192MB L2 (25.6MB as bf16) so the gather is
// L2-resident and the kernel is matrix-engine bound:
//   v_wmma_f32_16x16x32_bf16, f32 accumulation over K=256.
// W1 (64KB of packed bf16 B-fragments) is staged in LDS per workgroup via
// CDNA5 global_load_async_to_lds_b128 (ASYNCcnt); each wave grid-strides
// over 16-edge tiles reading B via conflict-free ds_load_b128 while the A
// gathers (LOADcnt) overlap with WMMA issue.
//
// NOTE: the LDS B reads are deliberately made opaque to LICM (per-tile
// asm-volatile zero folded into the index) — otherwise LLVM hoists all 64
// fragments (512 VGPRs) out of the tile loop and spills them to scratch.
// ---------------------------------------------------------------------------

#define N_NODES 100000
#define DDIM    128
#define N_EDGES 625000
#define NTILES  ((N_EDGES + 15) / 16)   // 39063 wave-tiles
#define GRID_BLOCKS 1024

typedef __attribute__((ext_vector_type(16))) __bf16 v16bf;
typedef __attribute__((ext_vector_type(8)))  float  v8f;

union Frag {
    uint4 u[2];   // 32 bytes
    v16bf v;      // 16 x bf16
};

// round-to-nearest-even float -> bf16 (bit pattern as ushort)
__device__ __forceinline__ unsigned short f2bf(float f) {
    unsigned int u = __float_as_uint(f);
    unsigned int lsb = (u >> 16) & 1u;
    u += 0x7fffu + lsb;
    return (unsigned short)(u >> 16);
}

// ---- prep 1: x (fp32) -> bf16 workspace ---------------------------------
__global__ void cvt_x_kernel(const float* __restrict__ x,
                             unsigned short* __restrict__ xb, int n4) {
    int i = blockIdx.x * blockDim.x + threadIdx.x;
    if (i >= n4) return;
    float4 f = reinterpret_cast<const float4*>(x)[i];
    ushort4 o;
    o.x = f2bf(f.x); o.y = f2bf(f.y); o.z = f2bf(f.z); o.w = f2bf(f.w);
    reinterpret_cast<ushort4*>(xb)[i] = o;
}

// ---- prep 2: pack W1 (256x128 row-major fp32) into bf16 B-fragments ----
// Plane-interleaved layout (for conflict-free ds_load_b128):
//   uint4 chunk index c = tile*64 + uh*32 + lane,  tile = n0*8+k0, uh = 0/1
//   chunk holds fragment elements e = uh*8 .. uh*8+7 for that lane.
// ISA 16-bit B layout: half = lane>>4, col = lane&15,
//   kk(e) = e<8 ? half*8+e : 16+half*8+(e-8)
//   element = W1[(k0*32+kk)*128 + (n0*16+col)]
__global__ void pack_w1_kernel(const float* __restrict__ W1,
                               unsigned short* __restrict__ w1p) {
    int id = blockIdx.x * blockDim.x + threadIdx.x;   // ushort index
    if (id >= 64 * 64 * 8) return;                    // 32768 elements
    int e7    = id & 7;
    int chunk = id >> 3;          // uint4 chunk
    int lane  = chunk & 31;
    int uh    = (chunk >> 5) & 1;
    int tile  = chunk >> 6;       // 0..63
    int k0    = tile & 7;
    int n0    = tile >> 3;
    int e     = uh * 8 + e7;
    int half  = lane >> 4;
    int col   = lane & 15;
    int kk    = (e < 8) ? (half * 8 + e) : (16 + half * 8 + (e - 8));
    w1p[id] = f2bf(W1[(k0 * 32 + kk) * DDIM + (n0 * 16 + col)]);
}

// ---- main kernel: 8 waves/block, each wave grid-strides over 16-edge tiles
__global__ void __launch_bounds__(256)
edge_mlp_kernel(const unsigned short* __restrict__ xb,
                const uint4*          __restrict__ w1p4,  // 4096 uint4 = 64KB
                const int*            __restrict__ ei,    // (E,2) int32
                const float*          __restrict__ b1,
                const float*          __restrict__ w2,
                const float*          __restrict__ b2,
                float*                __restrict__ out) {
    __shared__ uint4 lds_w1[4096];   // 64 KB of packed W1 fragments

    const int tid  = threadIdx.x;
    const int lane = tid & 31;
    const int wave = tid >> 5;
    const int half = lane >> 4;
    const int col  = lane & 15;

    // ---- stage W1 into LDS with CDNA5 async copies (ASYNCcnt) ----------
    #pragma unroll
    for (int it = 0; it < 16; ++it) {
        const int c = it * 256 + tid;                       // uint4 chunk
        const unsigned ldsoff = (unsigned)(size_t)(&lds_w1[c]);
        const unsigned long long gaddr =
            (unsigned long long)(size_t)(w1p4 + c);
        asm volatile("global_load_async_to_lds_b128 %0, %1, off"
                     :: "v"(ldsoff), "v"(gaddr) : "memory");
    }
    asm volatile("s_wait_asynccnt 0x0" ::: "memory");
    __syncthreads();

    // per-lane epilogue constants (column-indexed, tile-invariant)
    float bb[8], ww[8];
    #pragma unroll
    for (int n0 = 0; n0 < 8; ++n0) {
        bb[n0] = b1[n0 * 16 + col];
        ww[n0] = w2[n0 * 16 + col];
    }
    const float bias2 = b2[0];

    const long long tstride = (long long)gridDim.x * 8;
    for (long long t = (long long)blockIdx.x * 8 + wave; t < NTILES;
         t += tstride) {
        const long long base = t * 16;

        // Opaque zero, regenerated every tile: folded into the LDS index so
        // LICM cannot hoist the 128 ds_load_b128s (and spill) out of the loop.
        unsigned opq;
        asm volatile("v_mov_b32 %0, 0" : "=v"(opq));
        const int lidx = lane + (int)opq;

        // lane's A-row edge (clamped for the tail tile; stores are guarded)
        long long e = base + col;
        if (e >= N_EDGES) e = N_EDGES - 1;
        const int srcN = ei[2 * e + 0];
        const int tarN = ei[2 * e + 1];
        const unsigned short* rowS = xb + (size_t)srcN * DDIM;
        const unsigned short* rowT = xb + (size_t)tarN * DDIM;

        // preload ALL A fragments for the tile (one gather clause).
        // k0<4 -> src half of z, k0>=4 -> tar half; chunks are 16B-aligned
        // and never straddle the halves.
        Frag a[8];
        #pragma unroll
        for (int k0 = 0; k0 < 8; ++k0) {
            const unsigned short* r = (k0 < 4) ? rowS : rowT;
            const int o = (k0 & 3) * 32 + half * 8;
            a[k0].u[0] = *reinterpret_cast<const uint4*>(r + o);
            a[k0].u[1] = *reinterpret_cast<const uint4*>(r + o + 16);
        }

        v8f acc[8] = {};   // 8 N-tiles of 16x16 f32 accumulators
        #pragma unroll
        for (int k0 = 0; k0 < 8; ++k0) {
            #pragma unroll
            for (int n0 = 0; n0 < 8; ++n0) {
                Frag bfr;
                const int tb = (n0 * 8 + k0) * 64;
                bfr.u[0] = lds_w1[tb + lidx];        // conflict-free b128
                bfr.u[1] = lds_w1[tb + 32 + lidx];
                acc[n0] = __builtin_amdgcn_wmma_f32_16x16x32_bf16(
                    false, a[k0].v, false, bfr.v, (short)0, acc[n0],
                    false, false);
            }
        }

        // ---- fused epilogue: bias + relu + (128x1) GEMM + lane reduce --
        // C/D layout: VGPR v, lanes 0-15 -> M=v, lanes 16-31 -> M=v+8.
        float part[8];
        #pragma unroll
        for (int v = 0; v < 8; ++v) part[v] = 0.f;
        #pragma unroll
        for (int n0 = 0; n0 < 8; ++n0) {
            #pragma unroll
            for (int v = 0; v < 8; ++v) {
                float h = acc[n0][v] + bb[n0];
                h = h > 0.f ? h : 0.f;
                part[v] += h * ww[n0];
            }
        }
        #pragma unroll
        for (int v = 0; v < 8; ++v) {
            #pragma unroll
            for (int off = 8; off > 0; off >>= 1)
                part[v] += __shfl_xor(part[v], off, 16);
        }
        if (col == 0) {
            #pragma unroll
            for (int v = 0; v < 8; ++v) {
                const long long oe = base + half * 8 + v;
                if (oe < N_EDGES) out[oe] = part[v] + bias2;
            }
        }
    }
}

// ---------------------------------------------------------------------------
extern "C" void kernel_launch(void* const* d_in, const int* in_sizes, int n_in,
                              void* d_out, int out_size, void* d_ws, size_t ws_size,
                              hipStream_t stream) {
    const float* x  = (const float*)d_in[0];
    const int*   ei = (const int*)d_in[1];   // integer inputs arrive as int32
    const float* W1 = (const float*)d_in[2];
    const float* b1 = (const float*)d_in[3];
    const float* W2 = (const float*)d_in[4];
    const float* b2 = (const float*)d_in[5];
    float* out = (float*)d_out;

    // workspace: [ x_bf16 : 12.8M ushort = 25.6MB ][ W1 packed : 32K ushort ]
    unsigned short* xb  = (unsigned short*)d_ws;
    unsigned short* w1p = xb + (size_t)N_NODES * DDIM;   // 16B-aligned offset

    const int n4 = (N_NODES * DDIM) / 4;                 // 3.2M float4s
    cvt_x_kernel<<<(n4 + 255) / 256, 256, 0, stream>>>(x, xb, n4);
    pack_w1_kernel<<<(64 * 64 * 8) / 256, 256, 0, stream>>>(W1, w1p);

    edge_mlp_kernel<<<GRID_BLOCKS, 256, 0, stream>>>(
        xb, (const uint4*)w1p, ei, b1, W2, b2, out);
}
